// MultiHeadSelfAttention_15195594293659
// MI455X (gfx1250) — compile-verified, matrix operand order checked by reading, not run on hardware
//
#include <hip/hip_runtime.h>

typedef __bf16 v16bf __attribute__((ext_vector_type(16)));
typedef float  v8f   __attribute__((ext_vector_type(8)));
typedef unsigned int v4u __attribute__((ext_vector_type(4)));

static constexpr int Bsz = 2, T = 4096, C = 512, H = 8, Dh = 64;
static constexpr int Mrows = Bsz * T;    // 8192
static constexpr int N_qkv = 3 * C;      // 1536

__device__ __forceinline__ v8f zero8() {
  v8f z;
#pragma unroll
  for (int i = 0; i < 8; ++i) z[i] = 0.f;
  return z;
}

// Load a 16-element bf16 fragment from two contiguous 8-element (16B) chunks.
__device__ __forceinline__ v16bf load_frag(const __bf16* p0, const __bf16* p1) {
  union { v16bf v; v4u u[2]; } r;
  r.u[0] = *reinterpret_cast<const v4u*>(p0);
  r.u[1] = *reinterpret_cast<const v4u*>(p1);
  return r.v;
}

__device__ __forceinline__ v8f wmma_bf16(v16bf a, v16bf b, v8f c) {
  return __builtin_amdgcn_wmma_f32_16x16x32_bf16(false, a, false, b, (short)0, c,
                                                 false, false);
}

// ---------------- conversion kernels ----------------
__global__ void cvt_f32_bf16(const float* __restrict__ src,
                             __bf16* __restrict__ dst, int n) {
  int i = blockIdx.x * blockDim.x + threadIdx.x;
  if (i < n) dst[i] = (__bf16)src[i];
}

// src [rows, cols] fp32 -> dst [cols, rows] bf16  (dst[n][k] = src[k][n])
__global__ void cvt_transpose(const float* __restrict__ src,
                              __bf16* __restrict__ dst, int rows, int cols) {
  int i = blockIdx.x * blockDim.x + threadIdx.x;
  if (i < rows * cols) {
    int n = i / rows;
    int k = i - n * rows;
    dst[i] = (__bf16)src[(long)k * cols + n];
  }
}

// ---------------- WMMA GEMM:  out = A[M,K] * Bt[N,K]^T + bias ----------------
// One wave per 32x64 output tile (2 M-subtiles x 4 N-subtiles); every
// B-fragment is reused by two WMMAs. mode 0: scatter qkv into Q/K/Vt (bf16),
// mode 1: write fp32 rows (final projection).
__global__ __launch_bounds__(32) void gemm_wmma(
    const __bf16* __restrict__ A, const __bf16* __restrict__ Bt,
    const float* __restrict__ bias, int K, int mode, __bf16* __restrict__ Qb,
    __bf16* __restrict__ Kb, __bf16* __restrict__ Vtb,
    float* __restrict__ outF) {
  const int lane = threadIdx.x;
  const int half = lane >> 4;
  const int l15 = lane & 15;
  const int n_base = blockIdx.x * 64;
  const int m_base = blockIdx.y * 32;

  const __bf16* arow0 = A + (long)(m_base + l15) * K;
  const __bf16* arow1 = A + (long)(m_base + 16 + l15) * K;
  v8f acc[2][4];
#pragma unroll
  for (int mt = 0; mt < 2; ++mt)
#pragma unroll
    for (int nt = 0; nt < 4; ++nt) acc[mt][nt] = zero8();

  for (int k0 = 0; k0 < K; k0 += 32) {
    v16bf af0 = load_frag(arow0 + k0 + 8 * half, arow0 + k0 + 16 + 8 * half);
    v16bf af1 = load_frag(arow1 + k0 + 8 * half, arow1 + k0 + 16 + 8 * half);
#pragma unroll
    for (int nt = 0; nt < 4; ++nt) {
      const __bf16* brow = Bt + (long)(n_base + nt * 16 + l15) * K;
      v16bf bf = load_frag(brow + k0 + 8 * half, brow + k0 + 16 + 8 * half);
      acc[0][nt] = wmma_bf16(af0, bf, acc[0][nt]);
      acc[1][nt] = wmma_bf16(af1, bf, acc[1][nt]);
    }
  }

#pragma unroll
  for (int nt = 0; nt < 4; ++nt) {
    const int n = n_base + nt * 16 + l15;
    const float bv = bias[n];
#pragma unroll
    for (int mt = 0; mt < 2; ++mt) {
#pragma unroll
      for (int r = 0; r < 8; ++r) {
        const int m = m_base + mt * 16 + r + 8 * half;  // C-layout row
        const float val = acc[mt][nt][r] + bv;
        if (mode == 0) {
          const int which = n / C;
          const int cid = n % C;
          const int hh = cid / Dh;
          const int dd = cid % Dh;
          const int bb = m / T;
          const int t = m % T;
          const long bh = (long)(bb * H + hh);
          if (which == 0)
            Qb[(bh * T + t) * Dh + dd] = (__bf16)val;
          else if (which == 1)
            Kb[(bh * T + t) * Dh + dd] = (__bf16)val;
          else
            Vtb[(bh * Dh + dd) * T + t] = (__bf16)val;  // V stored transposed
        } else {
          outF[(long)m * C + n] = val;
        }
      }
    }
  }
}

// ---------------- flash attention, one wave per (b, h, 32-query tile) -------
// St = K_block(A, 16 keys x 32 d) * Q^T(B, 32 d x 16 q). C-layout of St
// (reg r = key, lane = query) maps in-lane onto the A-fragment layout
// (lane = query row, component c = key) needed by the PV WMMA. K/V fragment
// loads are shared across both 16-query subtiles.
__global__ __launch_bounds__(32) void attn_wmma(const __bf16* __restrict__ Qb,
                                                const __bf16* __restrict__ Kb,
                                                const __bf16* __restrict__ Vtb,
                                                __bf16* __restrict__ Obf) {
  const int lane = threadIdx.x;
  const int half = lane >> 4;
  const int l15 = lane & 15;
  const int q_base = blockIdx.x * 32;
  const int h = blockIdx.y;
  const int bb = blockIdx.z;
  const long bh = (long)(bb * H + h);
  const __bf16* Qp = Qb + bh * T * Dh;
  const __bf16* Kp = Kb + bh * T * Dh;
  const __bf16* Vp = Vtb + bh * Dh * T;

  // Q as B-operand: lane = query column, components = d (contiguous)
  v16bf qf[2][2];
#pragma unroll
  for (int qt = 0; qt < 2; ++qt) {
    const __bf16* qrow = Qp + (long)(q_base + qt * 16 + l15) * Dh;
    qf[qt][0] = load_frag(qrow + 8 * half, qrow + 16 + 8 * half);
    qf[qt][1] = load_frag(qrow + 32 + 8 * half, qrow + 48 + 8 * half);
  }

  v8f acc[2][4];
#pragma unroll
  for (int qt = 0; qt < 2; ++qt)
#pragma unroll
    for (int nt = 0; nt < 4; ++nt) acc[qt][nt] = zero8();
  float m_st[2] = {-3.0e38f, -3.0e38f};
  float l_st[2] = {0.f, 0.f};
  const float sc = 0.125f * 1.44269504088896340736f;  // 1/sqrt(D) * log2(e)
  const int q_my[2] = {q_base + l15, q_base + 16 + l15};
  const int nsteps = (q_base + 31) / 32 + 1;  // causal block skipping

  for (int j = 0; j < nsteps; ++j) {
    const int kb = j * 32;
    // K as A-operand: lane = key row, components = d
    const __bf16* kr0 = Kp + (long)(kb + l15) * Dh;
    const __bf16* kr1 = Kp + (long)(kb + 16 + l15) * Dh;
    v16bf ka[2][2];
    ka[0][0] = load_frag(kr0 + 8 * half, kr0 + 16 + 8 * half);
    ka[0][1] = load_frag(kr0 + 32 + 8 * half, kr0 + 48 + 8 * half);
    ka[1][0] = load_frag(kr1 + 8 * half, kr1 + 16 + 8 * half);
    ka[1][1] = load_frag(kr1 + 32 + 8 * half, kr1 + 48 + 8 * half);

    v16bf pa[2];  // P A-fragments, built entirely in-lane
#pragma unroll
    for (int qt = 0; qt < 2; ++qt) {
      v8f st0 = zero8(), st1 = zero8();
      st0 = wmma_bf16(ka[0][0], qf[qt][0], st0);
      st0 = wmma_bf16(ka[0][1], qf[qt][1], st0);
      st1 = wmma_bf16(ka[1][0], qf[qt][0], st1);
      st1 = wmma_bf16(ka[1][1], qf[qt][1], st1);

      // scale + causal mask; St element: key = kb + kt*16 + r + 8*half
      float mloc = -3.0e38f;
#pragma unroll
      for (int r = 0; r < 8; ++r) {
        const int k0i = kb + r + 8 * half;
        const int k1i = k0i + 16;
        const float s0 = (k0i <= q_my[qt]) ? st0[r] * sc : -3.0e38f;
        const float s1 = (k1i <= q_my[qt]) ? st1[r] * sc : -3.0e38f;
        st0[r] = s0;
        st1[r] = s1;
        mloc = fmaxf(mloc, fmaxf(s0, s1));
      }
      mloc = fmaxf(mloc, __shfl_xor(mloc, 16));
      const float m_new = fmaxf(m_st[qt], mloc);

      float rsum = 0.f;
#pragma unroll
      for (int r = 0; r < 8; ++r) {
        const float e0 = __builtin_amdgcn_exp2f(st0[r] - m_new);
        const float e1 = __builtin_amdgcn_exp2f(st1[r] - m_new);
        rsum += e0 + e1;
        pa[qt][r] = (__bf16)e0;
        pa[qt][r + 8] = (__bf16)e1;
      }
      rsum += __shfl_xor(rsum, 16);
      const float fac = __builtin_amdgcn_exp2f(m_st[qt] - m_new);
      l_st[qt] = l_st[qt] * fac + rsum;
      m_st[qt] = m_new;

#pragma unroll
      for (int r = 0; r < 8; ++r) {
        // O row for register r is q-subtile row (r + 8*half); its softmax
        // state lives in lane (r + 8*half) (lanes 0..15 hold q = lane).
        const float fr = __shfl(fac, r + 8 * half);
#pragma unroll
        for (int nt = 0; nt < 4; ++nt) acc[qt][nt][r] *= fr;
      }
    }

    // V as B-operand: lane = d column, components = keys (contiguous in Vt);
    // each V fragment feeds both query subtiles.
#pragma unroll
    for (int nt = 0; nt < 4; ++nt) {
      const __bf16* vcol = Vp + (long)(nt * 16 + l15) * T + kb;
      v16bf vf = load_frag(vcol + 8 * half, vcol + 16 + 8 * half);
      acc[0][nt] = wmma_bf16(pa[0], vf, acc[0][nt]);
      acc[1][nt] = wmma_bf16(pa[1], vf, acc[1][nt]);
    }
  }

  // finalize: divide by row sum, write bf16 [B, T, H*D] (== reshape B,T,C)
#pragma unroll
  for (int qt = 0; qt < 2; ++qt) {
#pragma unroll
    for (int r = 0; r < 8; ++r) {
      const float lr = __shfl(l_st[qt], r + 8 * half);
      const float inv = 1.0f / lr;
      const int t = q_base + qt * 16 + r + 8 * half;
#pragma unroll
      for (int nt = 0; nt < 4; ++nt) {
        const float val = acc[qt][nt][r] * inv;
        Obf[((long)(bb * T + t)) * C + h * Dh + nt * 16 + l15] = (__bf16)val;
      }
    }
  }
}

// ---------------------------------------------------------------------------
extern "C" void kernel_launch(void* const* d_in, const int* in_sizes, int n_in,
                              void* d_out, int out_size, void* d_ws,
                              size_t ws_size, hipStream_t stream) {
  (void)in_sizes; (void)n_in; (void)out_size; (void)ws_size;
  const float* x = (const float*)d_in[0];
  const float* Wqkv = (const float*)d_in[1];
  const float* bqkv = (const float*)d_in[2];
  const float* Wout = (const float*)d_in[3];
  const float* bout = (const float*)d_in[4];
  float* out = (float*)d_out;

  char* ws = (char*)d_ws;
  size_t off = 0;
  auto alloc = [&](size_t bytes) {
    void* p = ws + off;
    off += (bytes + 255) & ~(size_t)255;
    return p;
  };
  __bf16* xb  = (__bf16*)alloc((size_t)Mrows * C * 2);        // x in bf16
  __bf16* wqt = (__bf16*)alloc((size_t)N_qkv * C * 2);        // Wqkv^T bf16
  __bf16* wot = (__bf16*)alloc((size_t)C * C * 2);            // Wout^T bf16
  __bf16* Qb  = (__bf16*)alloc((size_t)Bsz * H * T * Dh * 2); // [B,H,T,D]
  __bf16* Kb  = (__bf16*)alloc((size_t)Bsz * H * T * Dh * 2); // [B,H,T,D]
  __bf16* Vtb = (__bf16*)alloc((size_t)Bsz * H * Dh * T * 2); // [B,H,D,T]
  __bf16* Obf = (__bf16*)alloc((size_t)Mrows * C * 2);        // attn out bf16

  const int n1 = Mrows * C;
  cvt_f32_bf16<<<(n1 + 255) / 256, 256, 0, stream>>>(x, xb, n1);
  const int n2 = C * N_qkv;
  cvt_transpose<<<(n2 + 255) / 256, 256, 0, stream>>>(Wqkv, wqt, C, N_qkv);
  const int n3 = C * C;
  cvt_transpose<<<(n3 + 255) / 256, 256, 0, stream>>>(Wout, wot, C, C);

  dim3 g1(N_qkv / 64, Mrows / 32);
  gemm_wmma<<<g1, 32, 0, stream>>>(xb, wqt, bqkv, C, 0, Qb, Kb, Vtb, nullptr);

  dim3 g2(T / 32, H, Bsz);
  attn_wmma<<<g2, 32, 0, stream>>>(Qb, Kb, Vtb, Obf);

  dim3 g3(C / 64, Mrows / 32);
  gemm_wmma<<<g3, 32, 0, stream>>>(Obf, wot, bout, C, 1, nullptr, nullptr,
                                   nullptr, out);
}
